// Feature_extract_23115513987551
// MI455X (gfx1250) — compile-verified, matrix operand order checked by reading, not run on hardware
//
#include <hip/hip_runtime.h>

typedef __attribute__((ext_vector_type(16))) _Float16 v16h;
typedef __attribute__((ext_vector_type(8)))  float    v8f;

#define BATCH 8
#define NPTS  4096
#define KNN   20
#define PE    (BATCH*NPTS*KNN)   // 655360 edge points
#define PN    (BATCH*NPTS)       // 32768 points
#define NEG_INF (-3.402823466e38f)
#define SLOPE 0.2f

__device__ __forceinline__ float lrelu(float v) { return v >= 0.f ? v : SLOPE * v; }

// ---------------------------------------------------------------------------
// kNN: one block per (b,n) query row. Distances in LDS, 20 rounds of argmax.
// ---------------------------------------------------------------------------
__global__ __launch_bounds__(256)
void knn_kernel(const float* __restrict__ pc, int* __restrict__ idx) {
    __shared__ float dist[NPTS];
    __shared__ float sval[256];
    __shared__ int   sidx[256];
    int tid = threadIdx.x;
    int bn  = blockIdx.x;
    int b   = bn >> 12, n = bn & 4095;
    const float* base = pc + (size_t)b * NPTS * 3;
    float cx = base[n*3+0], cy = base[n*3+1], cz = base[n*3+2];
    for (int m = tid; m < NPTS; m += 256) {
        float dx = base[m*3+0]-cx, dy = base[m*3+1]-cy, dz = base[m*3+2]-cz;
        dist[m] = -(dx*dx + dy*dy + dz*dz);
    }
    __syncthreads();
    for (int it = 0; it < KNN; ++it) {
        float bv = NEG_INF; int bi = 0x7fffffff;
        for (int m = tid; m < NPTS; m += 256) {
            float v = dist[m];
            if (v > bv || (v == bv && m < bi)) { bv = v; bi = m; }
        }
        sval[tid] = bv; sidx[tid] = bi;
        __syncthreads();
        for (int s = 128; s > 0; s >>= 1) {
            if (tid < s) {
                float ov = sval[tid+s]; int oi = sidx[tid+s];
                if (ov > sval[tid] || (ov == sval[tid] && oi < sidx[tid])) {
                    sval[tid] = ov; sidx[tid] = oi;
                }
            }
            __syncthreads();
        }
        if (tid == 0) { idx[(size_t)bn*KNN + it] = sidx[0]; dist[sidx[0]] = NEG_INF; }
        __syncthreads();
    }
}

// ---------------------------------------------------------------------------
// zero scratch floats
// ---------------------------------------------------------------------------
__global__ void zero_kernel(float* p, int n) {
    int i = blockIdx.x * 256 + threadIdx.x;
    if (i < n) p[i] = 0.f;
}

// ---------------------------------------------------------------------------
// Generic 64-output-channel conv over edge points (WMMA f32_16x16x32_f16).
// mode 0: features = [x_j - x_n, x_n] from pc (Cin=6, kchunks=1)
// mode 1: features = lrelu(bn(pre_in))      (Cin=64, kchunks=2)
// mode 2: features = [x1_j - x1_n, x1_n]    (Cin=128, kchunks=4)
// Block: 256 threads = 8 waves; each wave: private M-tile of 16 edge points.
// A fragments are built per-lane directly (ISA A-layout is lane-computable);
// only the weights are staged in LDS (shared by all 8 waves, staged once).
// Also accumulates per-channel sum / sumsq for BatchNorm (training stats).
// ---------------------------------------------------------------------------
__global__ __launch_bounds__(256)
void conv64_kernel(int mode, int kchunks, int Cin,
                   const float* __restrict__ W,          // [64, Cin]
                   const _Float16* __restrict__ pre_in,  // mode 1
                   const float* __restrict__ scale,      // mode 1 (BN scale)
                   const float* __restrict__ shift,      // mode 1 (BN shift)
                   const float* __restrict__ pcg,        // mode 0
                   const int*   __restrict__ idxg,       // mode 0,2
                   const float* __restrict__ x1g,        // mode 2
                   _Float16* __restrict__ preOut,        // [PE, 64]
                   float* __restrict__ gSum, float* __restrict__ gSq) {
    __shared__ _Float16 Btile[64][128];     // all K-chunks staged once (<=16KB)
    __shared__ float sSum[64], sSq[64];

    int tid  = threadIdx.x;
    int wv   = tid >> 5;
    int lane = tid & 31;
    int p0   = blockIdx.x * 128 + wv * 16;

    if (tid < 64) { sSum[tid] = 0.f; sSq[tid] = 0.f; }

    // --- stage all weights once (zero-padded to kchunks*32) ---
    int kw = kchunks * 32;
    for (int i = tid; i < 64 * kw; i += 256) {
        int n = i / kw, k = i - n * kw;
        Btile[n][k] = (_Float16)((k < Cin) ? W[(size_t)n*Cin + k] : 0.f);
    }
    __syncthreads();

    // --- per-lane A geometry (ISA 16-bit A 16x32 fragment layout) ---
    int row = lane & 15;             // M within tile
    int klo = (lane < 16) ? 0 : 8;   // K base for this half-wave
    int p   = p0 + row;              // edge point this lane's fragment row maps to
    int bn  = p / KNN, ke = p - bn * KNN;
    int b   = bn >> 12, n = bn & 4095;

    // mode-0 feature vector (only 6 valid channels)
    float f6[6];
    const float* xc = nullptr; const float* xj = nullptr;
    const _Float16* pin = nullptr;
    if (mode == 0) {
        int j = idxg[(size_t)bn*KNN + ke];
        const float* c3 = pcg + (size_t)(b*NPTS + n)*3;
        const float* n3 = pcg + (size_t)(b*NPTS + j)*3;
        f6[0]=n3[0]-c3[0]; f6[1]=n3[1]-c3[1]; f6[2]=n3[2]-c3[2];
        f6[3]=c3[0];       f6[4]=c3[1];       f6[5]=c3[2];
    } else if (mode == 1) {
        pin = pre_in + (size_t)p*64;
    } else {
        int j = idxg[(size_t)bn*KNN + ke];
        xc = x1g + (size_t)(b*NPTS + n)*64;
        xj = x1g + (size_t)(b*NPTS + j)*64;
    }

    v8f acc[4] = {};

    for (int kc = 0; kc < kchunks; ++kc) {
        // --- build A fragment directly (two contiguous 8-element segments) ---
        v16h afrag;
        if (mode == 0) {
            #pragma unroll
            for (int i = 0; i < 8; ++i) {
                int c0 = klo + i, c1 = klo + 16 + i;
                afrag[i]   = (_Float16)(c0 < 6 ? f6[c0] : 0.f);
                afrag[i+8] = (_Float16)(c1 < 6 ? f6[c1] : 0.f);
            }
        } else if (mode == 1) {
            int cb = kc*32 + klo;
            #pragma unroll
            for (int i = 0; i < 8; ++i) {
                int c0 = cb + i, c1 = cb + 16 + i;
                afrag[i]   = (_Float16)lrelu((float)pin[c0] * scale[c0] + shift[c0]);
                afrag[i+8] = (_Float16)lrelu((float)pin[c1] * scale[c1] + shift[c1]);
            }
        } else {
            int cb = kc*32 + klo;
            #pragma unroll
            for (int i = 0; i < 8; ++i) {
                int c0 = cb + i, c1 = cb + 16 + i;
                afrag[i]   = (_Float16)((c0 < 64) ? (xj[c0] - xc[c0]) : xc[c0-64]);
                afrag[i+8] = (_Float16)((c1 < 64) ? (xj[c1] - xc[c1]) : xc[c1-64]);
            }
        }
        // --- 4 output-channel tiles, B fragments from LDS ---
        #pragma unroll
        for (int nt = 0; nt < 4; ++nt) {
            v16h bfrag;
            const _Float16* bp = &Btile[nt*16 + row][kc*32 + klo];
            #pragma unroll
            for (int i = 0; i < 8; ++i) { bfrag[i] = bp[i]; bfrag[i+8] = bp[i+16]; }
            acc[nt] = __builtin_amdgcn_wmma_f32_16x16x32_f16(
                false, afrag, false, bfrag, (short)0, acc[nt], false, false);
        }
    }

    // --- store pre-activations + accumulate BN stats ---
    int m_lo = (lane < 16) ? 0 : 8;
    #pragma unroll
    for (int nt = 0; nt < 4; ++nt) {
        int ch = nt*16 + row;
        float s = 0.f, q = 0.f;
        #pragma unroll
        for (int r = 0; r < 8; ++r) {
            float v = acc[nt][r];
            preOut[(size_t)(p0 + r + m_lo)*64 + ch] = (_Float16)v;
            s += v; q += v*v;
        }
        atomicAdd(&sSum[ch], s);
        atomicAdd(&sSq[ch], q);
    }
    __syncthreads();
    if (tid < 64) { atomicAdd(&gSum[tid], sSum[tid]); atomicAdd(&gSq[tid], sSq[tid]); }
}

// ---------------------------------------------------------------------------
// Conv3: [PN,128] x W3^T[128,1024]. Block = one M-tile of 16 points shared by
// all 8 waves (LDS A-tile is genuine reuse here); each wave owns 128 output
// channels (8 accumulators). B fragments loaded straight from global W3.
// ---------------------------------------------------------------------------
__global__ __launch_bounds__(256)
void conv3_kernel(const float* __restrict__ x1g, const float* __restrict__ x2g,
                  const float* __restrict__ W3,
                  _Float16* __restrict__ pre3,
                  float* __restrict__ gSum, float* __restrict__ gSq) {
    __shared__ _Float16 A3[16][128];
    __shared__ float sSum[1024], sSq[1024];
    int tid  = threadIdx.x;
    int wv   = tid >> 5;
    int lane = tid & 31;
    int p0   = blockIdx.x * 16;

    for (int i = tid; i < 1024; i += 256) { sSum[i] = 0.f; sSq[i] = 0.f; }
    for (int i = tid; i < 16*128; i += 256) {
        int r = i >> 7, c = i & 127;
        size_t p = (size_t)(p0 + r);
        float v = (c < 64) ? x1g[p*64 + c] : x2g[p*64 + (c-64)];
        A3[r][c] = (_Float16)v;
    }
    __syncthreads();

    v8f acc[8] = {};
    #pragma unroll
    for (int kc = 0; kc < 4; ++kc) {
        v16h afrag;
        {
            int row = lane & 15;
            int kb  = kc*32 + ((lane < 16) ? 0 : 8);
            const _Float16* ap = &A3[row][kb];
            #pragma unroll
            for (int i = 0; i < 8; ++i) { afrag[i] = ap[i]; afrag[i+8] = ap[i+16]; }
        }
        #pragma unroll
        for (int nt = 0; nt < 8; ++nt) {
            int ch = wv*128 + nt*16 + (lane & 15);
            int kb = kc*32 + ((lane < 16) ? 0 : 8);
            const float* bp = W3 + (size_t)ch*128 + kb;
            v16h bfrag;
            #pragma unroll
            for (int i = 0; i < 8; ++i) { bfrag[i] = (_Float16)bp[i]; bfrag[i+8] = (_Float16)bp[i+16]; }
            acc[nt] = __builtin_amdgcn_wmma_f32_16x16x32_f16(
                false, afrag, false, bfrag, (short)0, acc[nt], false, false);
        }
    }

    int m_lo = (lane < 16) ? 0 : 8;
    #pragma unroll
    for (int nt = 0; nt < 8; ++nt) {
        int ch = wv*128 + nt*16 + (lane & 15);
        float s = 0.f, q = 0.f;
        #pragma unroll
        for (int r = 0; r < 8; ++r) {
            float v = acc[nt][r];
            pre3[(size_t)(p0 + r + m_lo)*1024 + ch] = (_Float16)v;
            s += v; q += v*v;
        }
        atomicAdd(&sSum[ch], s);
        atomicAdd(&sSq[ch], q);
    }
    __syncthreads();
    for (int i = tid; i < 1024; i += 256) {
        atomicAdd(&gSum[i], sSum[i]); atomicAdd(&gSq[i], sSq[i]);
    }
}

// ---------------------------------------------------------------------------
// BN stats -> per-channel (scale, shift)
// ---------------------------------------------------------------------------
__global__ void bn_finalize_kernel(const float* gSum, const float* gSq,
                                   const float* gamma, const float* beta,
                                   float* scale, float* shift,
                                   int nch, float inv_cnt) {
    int c = blockIdx.x * 256 + threadIdx.x;
    if (c < nch) {
        float m = gSum[c] * inv_cnt;
        float v = gSq[c] * inv_cnt - m*m;
        float r = rsqrtf(v + 1e-5f);
        float sc = r * gamma[c];
        scale[c] = sc;
        shift[c] = beta[c] - m*sc;
    }
}

// ---------------------------------------------------------------------------
// BN + lrelu + max over K=20; writes channel-last x buffer and x_per output.
// ---------------------------------------------------------------------------
__global__ __launch_bounds__(256)
void maxk_kernel(const _Float16* __restrict__ pre,
                 const float* __restrict__ scale, const float* __restrict__ shift,
                 float* __restrict__ x_out, float* __restrict__ out_xper, int c_off) {
    int g = blockIdx.x * 256 + threadIdx.x;   // < PN*64
    int c  = g & 63;
    int bn = g >> 6;
    float sc = scale[c], sh = shift[c];
    float mx = NEG_INF;
    for (int k = 0; k < KNN; ++k) {
        float v = lrelu((float)pre[((size_t)bn*KNN + k)*64 + c] * sc + sh);
        mx = fmaxf(mx, v);
    }
    x_out[(size_t)bn*64 + c] = mx;
    int b = bn >> 12, n = bn & 4095;
    out_xper[((size_t)b*128 + c_off + c)*NPTS + n] = mx;
}

// ---------------------------------------------------------------------------
// BN + lrelu + max over N=4096 -> x_global [B,1024]
// Strided f16 walk: software prefetch (global_prefetch_b8) to hide HBM latency.
// ---------------------------------------------------------------------------
__global__ __launch_bounds__(256)
void maxn_kernel(const _Float16* __restrict__ pre3,
                 const float* __restrict__ scale, const float* __restrict__ shift,
                 float* __restrict__ xg) {
    int b = blockIdx.x >> 2;
    int c = (blockIdx.x & 3) * 256 + threadIdx.x;
    float sc = scale[c], sh = shift[c];
    const _Float16* col = pre3 + (size_t)b*NPTS*1024 + c;
    float mx = NEG_INF;
    for (int n = 0; n < NPTS; ++n) {
        if (n + 16 < NPTS) __builtin_prefetch(col + (size_t)(n+16)*1024, 0, 1);
        float v = lrelu((float)col[(size_t)n*1024] * sc + sh);
        mx = fmaxf(mx, v);
    }
    xg[b*1024 + c] = mx;
}

// ---------------------------------------------------------------------------
// VAE head: mu, log_var, z
// ---------------------------------------------------------------------------
__global__ __launch_bounds__(256)
void head_kernel(const float* __restrict__ xg,
                 const float* __restrict__ Wmu, const float* __restrict__ bmu,
                 const float* __restrict__ Wvar, const float* __restrict__ bvar,
                 const float* __restrict__ eps,
                 float* __restrict__ z_ws, float* __restrict__ out_z,
                 float* __restrict__ out_mu, float* __restrict__ out_lv) {
    int b = blockIdx.x, t = threadIdx.x;     // t in [0,256)
    const float* x  = xg + (size_t)b*1024;
    const float* wm = Wmu  + (size_t)t*1024;
    const float* wl = Wvar + (size_t)t*1024;
    float mu = bmu[t], lv = bvar[t];
    for (int c = 0; c < 1024; ++c) { float xv = x[c]; mu += xv*wm[c]; lv += xv*wl[c]; }
    float z = eps[b*256 + t] * expf(0.5f*lv) + mu;
    out_mu[b*256 + t] = mu;
    out_lv[b*256 + t] = lv;
    out_z [b*256 + t] = z;
    z_ws  [b*256 + t] = z;
}

// ---------------------------------------------------------------------------
// Decoder: enc = lrelu(Wenc); xc = lrelu(Wc2 @ lrelu(Wc1 @ [z;enc]))
// ---------------------------------------------------------------------------
__global__ __launch_bounds__(256)
void decoder_kernel(const float* __restrict__ z,
                    const float* __restrict__ Wenc,
                    const float* __restrict__ Wc1,
                    const float* __restrict__ Wc2,
                    float* __restrict__ out_xc) {
    __shared__ float enc[64][16];
    __shared__ float h1[256][16];
    int b = blockIdx.x, t = threadIdx.x;
    for (int i = t; i < 64*16; i += 256) {
        int e = i >> 4, m = i & 15;
        enc[e][m] = lrelu(Wenc[e*16 + m]);
    }
    __syncthreads();
    // layer 1: thread t = output channel o in [0,256)
    {
        const float* zb = z + (size_t)b*256;
        const float* w1 = Wc1 + (size_t)t*320;
        float zacc = 0.f;
        for (int c = 0; c < 256; ++c) zacc += w1[c]*zb[c];
        float hm[16];
        #pragma unroll
        for (int m = 0; m < 16; ++m) hm[m] = zacc;
        for (int e = 0; e < 64; ++e) {
            float w = w1[256 + e];
            #pragma unroll
            for (int m = 0; m < 16; ++m) hm[m] += w*enc[e][m];
        }
        #pragma unroll
        for (int m = 0; m < 16; ++m) h1[t][m] = lrelu(hm[m]);
    }
    __syncthreads();
    // layer 2: threads 0..127 = output channel o2
    if (t < 128) {
        const float* w2 = Wc2 + (size_t)t*256;
        float om[16];
        #pragma unroll
        for (int m = 0; m < 16; ++m) om[m] = 0.f;
        for (int c = 0; c < 256; ++c) {
            float w = w2[c];
            #pragma unroll
            for (int m = 0; m < 16; ++m) om[m] += w*h1[c][m];
        }
        #pragma unroll
        for (int m = 0; m < 16; ++m)
            out_xc[((size_t)b*128 + t)*16 + m] = lrelu(om[m]);
    }
}

// ---------------------------------------------------------------------------
extern "C" void kernel_launch(void* const* d_in, const int* in_sizes, int n_in,
                              void* d_out, int out_size, void* d_ws, size_t ws_size,
                              hipStream_t stream) {
    (void)in_sizes; (void)n_in; (void)out_size; (void)ws_size;
    const float* pc  = (const float*)d_in[0];
    const float* eps = (const float*)d_in[1];
    const float* W1a = (const float*)d_in[2];
    const float* g1a = (const float*)d_in[3];
    const float* b1a = (const float*)d_in[4];
    const float* W1b = (const float*)d_in[5];
    const float* g1b = (const float*)d_in[6];
    const float* b1b = (const float*)d_in[7];
    const float* W2a = (const float*)d_in[8];
    const float* g2a = (const float*)d_in[9];
    const float* b2a = (const float*)d_in[10];
    const float* W2b = (const float*)d_in[11];
    const float* g2b = (const float*)d_in[12];
    const float* b2b = (const float*)d_in[13];
    const float* W3  = (const float*)d_in[14];
    const float* g3  = (const float*)d_in[15];
    const float* b3  = (const float*)d_in[16];
    const float* Wmu = (const float*)d_in[17];
    const float* bmu = (const float*)d_in[18];
    const float* Wvar= (const float*)d_in[19];
    const float* bvar= (const float*)d_in[20];
    const float* Wenc= (const float*)d_in[21];
    const float* Wc1 = (const float*)d_in[22];
    const float* Wc2 = (const float*)d_in[23];

    // ---- workspace layout ----
    char* ws = (char*)d_ws;
    size_t off = 0;
    auto take = [&](size_t bytes) { size_t o = off; off += (bytes + 255) & ~(size_t)255; return o; };
    int*       idxp  = (int*)      (ws + take((size_t)PE * 4));
    _Float16*  preA  = (_Float16*) (ws + take((size_t)PE * 64 * 2));
    _Float16*  preB  = (_Float16*) (ws + take((size_t)PE * 64 * 2));
    float*     x1p   = (float*)    (ws + take((size_t)PN * 64 * 4));
    float*     x2p   = (float*)    (ws + take((size_t)PN * 64 * 4));
    float*     xgp   = (float*)    (ws + take((size_t)BATCH * 1024 * 4));
    float*     zp    = (float*)    (ws + take((size_t)BATCH * 256 * 4));
    float*     gSum  = (float*)    (ws + take(1024 * 4));
    float*     gSq   = (float*)    (ws + take(1024 * 4));
    float*     bnSc  = (float*)    (ws + take(1024 * 4));
    float*     bnSh  = (float*)    (ws + take(1024 * 4));
    _Float16*  pre3p = preA;   // aliased: preA (84MB) is dead by conv3 time (needs 67MB)

    // ---- output layout (tuple flattened) ----
    float* out      = (float*)d_out;
    float* out_xper = out;                                        // [8,128,4096]
    float* out_xc   = out_xper + (size_t)BATCH*128*NPTS;          // [8,128,16]
    float* out_z    = out_xc   + (size_t)BATCH*128*16;            // [8,256,1]
    float* out_mu   = out_z    + (size_t)BATCH*256;               // [8,256]
    float* out_lv   = out_mu   + (size_t)BATCH*256;               // [8,256]

    const float invPE = 1.0f / (float)PE;
    const float invPN = 1.0f / (float)PN;

    // kNN
    knn_kernel<<<PN, 256, 0, stream>>>(pc, idxp);

    // EdgeConv1 a
    zero_kernel<<<8, 256, 0, stream>>>(gSum, 2048);   // gSum+gSq contiguous
    conv64_kernel<<<PE/128, 256, 0, stream>>>(0, 1, 6, W1a, nullptr, nullptr, nullptr,
                                              pc, idxp, nullptr, preA, gSum, gSq);
    bn_finalize_kernel<<<1, 256, 0, stream>>>(gSum, gSq, g1a, b1a, bnSc, bnSh, 64, invPE);
    // EdgeConv1 b (BN1a+lrelu applied on load)
    zero_kernel<<<8, 256, 0, stream>>>(gSum, 2048);
    conv64_kernel<<<PE/128, 256, 0, stream>>>(1, 2, 64, W1b, preA, bnSc, bnSh,
                                              nullptr, nullptr, nullptr, preB, gSum, gSq);
    bn_finalize_kernel<<<1, 256, 0, stream>>>(gSum, gSq, g1b, b1b, bnSc, bnSh, 64, invPE);
    maxk_kernel<<<(PN*64)/256, 256, 0, stream>>>(preB, bnSc, bnSh, x1p, out_xper, 0);

    // EdgeConv2 a (features gathered from x1)
    zero_kernel<<<8, 256, 0, stream>>>(gSum, 2048);
    conv64_kernel<<<PE/128, 256, 0, stream>>>(2, 4, 128, W2a, nullptr, nullptr, nullptr,
                                              nullptr, idxp, x1p, preA, gSum, gSq);
    bn_finalize_kernel<<<1, 256, 0, stream>>>(gSum, gSq, g2a, b2a, bnSc, bnSh, 64, invPE);
    // EdgeConv2 b
    zero_kernel<<<8, 256, 0, stream>>>(gSum, 2048);
    conv64_kernel<<<PE/128, 256, 0, stream>>>(1, 2, 64, W2b, preA, bnSc, bnSh,
                                              nullptr, nullptr, nullptr, preB, gSum, gSq);
    bn_finalize_kernel<<<1, 256, 0, stream>>>(gSum, gSq, g2b, b2b, bnSc, bnSh, 64, invPE);
    maxk_kernel<<<(PN*64)/256, 256, 0, stream>>>(preB, bnSc, bnSh, x2p, out_xper, 64);

    // Global feature conv (1024 ch)
    zero_kernel<<<8, 256, 0, stream>>>(gSum, 2048);
    conv3_kernel<<<PN/16, 256, 0, stream>>>(x1p, x2p, W3, pre3p, gSum, gSq);
    bn_finalize_kernel<<<4, 256, 0, stream>>>(gSum, gSq, g3, b3, bnSc, bnSh, 1024, invPN);
    maxn_kernel<<<BATCH*4, 256, 0, stream>>>(pre3p, bnSc, bnSh, xgp);

    // VAE head + decoder
    head_kernel<<<BATCH, 256, 0, stream>>>(xgp, Wmu, bmu, Wvar, bvar, eps,
                                           zp, out_z, out_mu, out_lv);
    decoder_kernel<<<BATCH, 256, 0, stream>>>(zp, Wenc, Wc1, Wc2, out_xc);
}
